// encoder_RNN_62740882260823
// MI455X (gfx1250) — matrix operand order repaired and checked by reading, not compiled, not run on hardware
//
#include <hip/hip_runtime.h>

// ---------------------------------------------------------------------------
// MI455X (gfx1250) fused attention-LSTM encoder.
//
// Math simplification (exact): the per-row scalar (h@wh + c@wc + b_lin) is
// constant across the softmax axis, so a = softmax(s_x, axis=1) is
// time-invariant. Per step only:  gates = (a*x_t)@W_ih^T + h@W_hh^T + bias.
//
// Design:
//  * Weight B-fragments hoisted into registers (loop-invariant over the 128
//    serial timesteps) -> no per-step weight ds_loads / dscnt stalls.
//  * 58KB static LDS (x_t double buffer, gate exchange, bf16 A tiles, bias).
//  * Double-buffered async global->LDS prefetch of the next x_t tile
//    (global_load_async_to_lds_b128 / s_wait_asynccnt): the int4* AS1/AS3
//    pointer types match the builtin's prototype per the round-2 diagnostic.
// ---------------------------------------------------------------------------

typedef __attribute__((ext_vector_type(16))) __bf16 v16bf;
typedef __attribute__((ext_vector_type(8)))  float  v8f;
typedef __attribute__((ext_vector_type(4)))  int    v4i;

#define AS1 __attribute__((address_space(1)))
#define AS3 __attribute__((address_space(3)))

#define B_SZ    2048
#define T_STEPS 128
#define F_SZ    128
#define H_SZ    128
#define G_SZ    512      // 4*H
#define K_TOT   256      // F + H (combined GEMM K)
#define BTILE   16       // batch rows per block

__device__ __forceinline__ unsigned short f2bf(float f) {
  unsigned int u = __float_as_uint(f);
  unsigned int r = (u + 0x7FFFu + ((u >> 16) & 1u)) >> 16;   // round-to-nearest-even
  return (unsigned short)r;
}
__device__ __forceinline__ float sigf(float x) { return 1.0f / (1.0f + __expf(-x)); }
__device__ __forceinline__ float tanh_fast(float x) {
  x = fminf(fmaxf(x, -15.0f), 15.0f);
  float e = __expf(2.0f * x);
  return (e - 1.0f) / (e + 1.0f);
}

// ---- CDNA5 async global->LDS copy (ASYNCcnt), with safe fallback ----------
__device__ __forceinline__ void async_b128_to_lds(const float* gsrc, float* ldst) {
#if __has_builtin(__builtin_amdgcn_global_load_async_to_lds_b128)
  __builtin_amdgcn_global_load_async_to_lds_b128(
      (AS1 v4i*)gsrc, (AS3 v4i*)ldst, /*offset=*/0, /*cpol=*/0);
#else
  *(uint4*)ldst = *(const uint4*)gsrc;
#endif
}
__device__ __forceinline__ void wait_async0() {
#if __has_builtin(__builtin_amdgcn_s_wait_asynccnt)
  __builtin_amdgcn_s_wait_asynccnt(0);
#endif
}

// ---------------------------------------------------------------------------
// Kernel 1: a[b,f] = softmax_f( sum_t x[b,t,f] * wx[t] );  wx = w_lin[0,256:384]
// ---------------------------------------------------------------------------
__global__ void __launch_bounds__(128)
attn_softmax_kernel(const float* __restrict__ x,
                    const float* __restrict__ w_lin,
                    float* __restrict__ a_out) {
  __shared__ float wx_s[T_STEPS];
  __shared__ float red[F_SZ];
  const int b = blockIdx.x;
  const int f = threadIdx.x;

  wx_s[f] = w_lin[2 * H_SZ + f];
  __syncthreads();

  const float* xb = x + (size_t)b * T_STEPS * F_SZ + f;
  float s = 0.0f;
#pragma unroll 4
  for (int t = 0; t < T_STEPS; ++t) s = fmaf(xb[(size_t)t * F_SZ], wx_s[t], s);

  red[f] = s;
  __syncthreads();
  for (int off = F_SZ / 2; off > 0; off >>= 1) {
    if (f < off) red[f] = fmaxf(red[f], red[f + off]);
    __syncthreads();
  }
  const float m = red[0];
  __syncthreads();
  const float e = __expf(s - m);
  red[f] = e;
  __syncthreads();
  for (int off = F_SZ / 2; off > 0; off >>= 1) {
    if (f < off) red[f] += red[f + off];
    __syncthreads();
  }
  a_out[(size_t)b * F_SZ + f] = e / red[0];
}

// ---------------------------------------------------------------------------
// Kernel 2: pack weights (bf16, N-major K-contiguous) + fused bias.
//   WcT[n][k] = bf16( k<128 ? w_ih[n,k] : w_hh[n,k-128] )
// ---------------------------------------------------------------------------
__global__ void __launch_bounds__(256)
prep_weights_kernel(const float* __restrict__ w_ih,
                    const float* __restrict__ w_hh,
                    const float* __restrict__ b_ih,
                    const float* __restrict__ b_hh,
                    unsigned short* __restrict__ wct,
                    float* __restrict__ bias) {
  const int n = blockIdx.x;    // 0..511
  const int k = threadIdx.x;   // 0..255
  const float v = (k < H_SZ) ? w_ih[n * F_SZ + k] : w_hh[n * H_SZ + (k - H_SZ)];
  wct[n * K_TOT + k] = f2bf(v);
  if (k == 0) bias[n] = b_ih[n] + b_hh[n];
}

// ---------------------------------------------------------------------------
// Kernel 3: persistent recurrent kernel. 128 blocks x 256 threads (8 waves),
// BTILE=16 batch rows per block. Weight fragments live in VGPRs for all 128
// steps; x_t tiles are async-prefetched into a double-buffered LDS tile.
// ---------------------------------------------------------------------------
__global__ void __launch_bounds__(256)
lstm_recurrent_kernel(const float* __restrict__ x,
                      const float* __restrict__ a_glob,
                      const unsigned short* __restrict__ wct_glob,
                      const float* __restrict__ bias_glob,
                      float* __restrict__ out_atn,
                      float* __restrict__ out_enc) {
  __shared__ float          s_xf[2][BTILE][F_SZ];     // 16 KB: async x_t double buffer
  __shared__ float          s_gate[BTILE][G_SZ];      // 32 KB
  __shared__ unsigned short s_xa[BTILE][F_SZ];        //  4 KB: bf16 A tile (xa)
  __shared__ unsigned short s_h[BTILE][H_SZ];         //  4 KB: bf16 A tile (h)
  __shared__ float          s_bias[G_SZ];             //  2 KB

  const int tid = threadIdx.x;
  const int b0  = blockIdx.x * BTILE;

  // wave mapping (wave32): wave owns N columns [wave*64, wave*64+64)
  const int wave = tid >> 5;
  const int lane = tid & 31;
  const int frow = lane & 15;   // A row / B column within a 16x16 tile
  const int sel  = lane >> 4;   // half-wave select (K sub-layout)

  // phase1 / phase3 mapping: row = tid/16 (0..15), 8-wide chunk = (tid%16)*8
  const int p_row = tid >> 4;
  const int p_c0  = (tid & 15) * 8;

  // ---- one-time: bias to LDS, zero h tile ----
  for (int i = tid; i < G_SZ; i += 256) s_bias[i] = bias_glob[i];
  {
    uint4 z = make_uint4(0, 0, 0, 0);
    uint4* hz = (uint4*)&s_h[0][0];
    for (int i = tid; i < (BTILE * H_SZ * 2) / 16; i += 256) hz[i] = z;
  }

  // ---- one-time: hoist all B (weight) fragments into registers.
  // Per wave: 8 K-tiles x 4 N-tiles x 32B = 1 KB/lane = 256 VGPRs, held for
  // the whole serial loop (gfx1250 supports up to 1024 VGPRs/wave).
  union BFrag { uint4 u[2]; v16bf v; };
  BFrag wreg[8][4];
#pragma unroll
  for (int kt = 0; kt < 8; ++kt) {
#pragma unroll
    for (int j = 0; j < 4; ++j) {
      const unsigned short* bp =
          wct_glob + (wave * 64 + j * 16 + frow) * K_TOT + kt * 32 + sel * 16;
      wreg[kt][j].u[0] = *(const uint4*)(bp);
      wreg[kt][j].u[1] = *(const uint4*)(bp + 8);
    }
  }

  float a_reg[8];
#pragma unroll
  for (int e = 0; e < 8; ++e)
    a_reg[e] = a_glob[(size_t)(b0 + p_row) * F_SZ + p_c0 + e];

  float c_reg[8];
#pragma unroll
  for (int e = 0; e < 8; ++e) c_reg[e] = 0.0f;

  // ---- async prefetch of x tile for t=0: wave w loads rows 2w, 2w+1;
  //      each row is 512B = one b128 per lane.
  const int ar0 = wave * 2;
  {
#pragma unroll
    for (int r = 0; r < 2; ++r)
      async_b128_to_lds(x + ((size_t)(b0 + ar0 + r) * T_STEPS + 0) * F_SZ + lane * 4,
                        &s_xf[0][ar0 + r][lane * 4]);
  }
  __syncthreads();

  for (int t = 0; t < T_STEPS; ++t) {
    const int buf = t & 1;

    // current tile complete (each wave waits for its own async issues, then
    // the barrier publishes every wave's rows to the whole workgroup)
    wait_async0();
    __syncthreads();

    // kick off next tile into the other buffer; latency hides behind
    // phases 1-3 of this iteration.
    if (t + 1 < T_STEPS) {
#pragma unroll
      for (int r = 0; r < 2; ++r)
        async_b128_to_lds(
            x + ((size_t)(b0 + ar0 + r) * T_STEPS + (t + 1)) * F_SZ + lane * 4,
            &s_xf[buf ^ 1][ar0 + r][lane * 4]);
    }

    // ---------------- phase 1: xa = a * x_t ----------------
    {
      const float* xp = &s_xf[buf][p_row][p_c0];
      float xv[8];
#pragma unroll
      for (int e = 0; e < 8; ++e) xv[e] = xp[e] * a_reg[e];

      float* op = out_atn + ((size_t)(b0 + p_row) * T_STEPS + t) * F_SZ + p_c0;
#pragma unroll
      for (int e = 0; e < 8; ++e) op[e] = xv[e];

      uint4 pk;
      pk.x = (unsigned)f2bf(xv[0]) | ((unsigned)f2bf(xv[1]) << 16);
      pk.y = (unsigned)f2bf(xv[2]) | ((unsigned)f2bf(xv[3]) << 16);
      pk.z = (unsigned)f2bf(xv[4]) | ((unsigned)f2bf(xv[5]) << 16);
      pk.w = (unsigned)f2bf(xv[6]) | ((unsigned)f2bf(xv[7]) << 16);
      *(uint4*)(&s_xa[p_row][p_c0]) = pk;
    }
    __syncthreads();

    // ---------------- phase 2: gates = [xa|h] @ W^T (weights in VGPRs) -----
    {
      v8f acc[4] = {};
#pragma unroll
      for (int kt = 0; kt < 8; ++kt) {
        const unsigned short* atile = (kt < 4) ? &s_xa[0][0] : &s_h[0][0];
        const int kb = (kt & 3) * 32;
        union { uint4 u[2]; v16bf v; } af;
        const unsigned short* ap = atile + frow * F_SZ + kb + sel * 16;
        af.u[0] = *(const uint4*)(ap);        // K = kb + sel*16 + [0..7]
        af.u[1] = *(const uint4*)(ap + 8);    // K = kb + sel*16 + 8 + [0..7]
#pragma unroll
        for (int j = 0; j < 4; ++j)
          acc[j] = __builtin_amdgcn_wmma_f32_16x16x32_bf16(
              false, af.v, false, wreg[kt][j].v, (short)0, acc[j], false, false);
      }
      // C/D layout: VGPR r, lanes 0-15 -> M=r, lanes 16-31 -> M=8+r; N=lane%16
#pragma unroll
      for (int j = 0; j < 4; ++j) {
        const int n0 = wave * 64 + j * 16;
#pragma unroll
        for (int r = 0; r < 8; ++r)
          s_gate[sel * 8 + r][n0 + frow] = acc[j][r];
      }
    }
    __syncthreads();

    // ---------------- phase 3: LSTM cell update ----------------
    {
      const int b  = p_row;
      const int h0 = p_c0;
      float hnew[8];
#pragma unroll
      for (int e = 0; e < 8; ++e) {
        const int h = h0 + e;
        const float gi = s_gate[b][h]       + s_bias[h];
        const float gf = s_gate[b][128 + h] + s_bias[128 + h];
        const float gg = s_gate[b][256 + h] + s_bias[256 + h];
        const float go = s_gate[b][384 + h] + s_bias[384 + h];
        const float cn = sigf(gf) * c_reg[e] + sigf(gi) * tanh_fast(gg);
        c_reg[e] = cn;
        hnew[e]  = sigf(go) * tanh_fast(cn);
      }
      float* ep = out_enc + ((size_t)(b0 + b) * T_STEPS + t) * H_SZ + h0;
#pragma unroll
      for (int e = 0; e < 8; ++e) ep[e] = hnew[e];

      uint4 pk;
      pk.x = (unsigned)f2bf(hnew[0]) | ((unsigned)f2bf(hnew[1]) << 16);
      pk.y = (unsigned)f2bf(hnew[2]) | ((unsigned)f2bf(hnew[3]) << 16);
      pk.z = (unsigned)f2bf(hnew[4]) | ((unsigned)f2bf(hnew[5]) << 16);
      pk.w = (unsigned)f2bf(hnew[6]) | ((unsigned)f2bf(hnew[7]) << 16);
      *(uint4*)(&s_h[b][h0]) = pk;
    }
    __syncthreads();
  }
}

// ---------------------------------------------------------------------------
// Launcher
// ---------------------------------------------------------------------------
extern "C" void kernel_launch(void* const* d_in, const int* in_sizes, int n_in,
                              void* d_out, int out_size, void* d_ws, size_t ws_size,
                              hipStream_t stream) {
  (void)in_sizes; (void)n_in; (void)out_size; (void)ws_size;

  const float* x     = (const float*)d_in[0];   // (2048,128,128)
  const float* w_lin = (const float*)d_in[1];   // (1,384)
  // d_in[2] = b_lin: mathematically cancels in the softmax -> unused
  const float* w_ih  = (const float*)d_in[3];   // (512,128)
  const float* w_hh  = (const float*)d_in[4];   // (512,128)
  const float* b_ih  = (const float*)d_in[5];   // (512,)
  const float* b_hh  = (const float*)d_in[6];   // (512,)

  float* out      = (float*)d_out;
  float* out_atn  = out;                                          // (B,T,F)
  float* out_enc  = out + (size_t)B_SZ * T_STEPS * F_SZ;          // (B,T,H)

  // workspace layout: a (1MB f32) | WcT (256KB bf16) | bias (2KB f32)
  float*          ws_a    = (float*)d_ws;
  unsigned short* ws_w    = (unsigned short*)(ws_a + (size_t)B_SZ * F_SZ);
  float*          ws_bias = (float*)(ws_w + (size_t)G_SZ * K_TOT);

  attn_softmax_kernel<<<B_SZ, F_SZ, 0, stream>>>(x, w_lin, ws_a);
  prep_weights_kernel<<<G_SZ, K_TOT, 0, stream>>>(w_ih, w_hh, b_ih, b_hh, ws_w, ws_bias);
  lstm_recurrent_kernel<<<B_SZ / BTILE, 256, 0, stream>>>(
      x, ws_a, ws_w, ws_bias, out_atn, out_enc);
}
